// HeteroInteractionBlock_2877628088816
// MI455X (gfx1250) — matrix operand order repaired and checked by reading, not compiled, not run on hardware
//
#include <hip/hip_runtime.h>
#include <hip/hip_bf16.h>
#include <math.h>

typedef __attribute__((ext_vector_type(16))) _Float16 v16h;
typedef __attribute__((ext_vector_type(8)))  float    v8f;

#define WMMA16(a, b, c) \
  __builtin_amdgcn_wmma_f32_16x16x32_f16(false, (a), false, (b), (short)0, (c), false, false)

// Per-wave LDS write->read ordering (DS ops are in-order within a wave).
#define LDS_FENCE() __asm__ __volatile__("s_wait_dscnt 0" ::: "memory")

// K index pattern for 16-bit A fragments (16x32): lanes 0-15 hold K {0..7,16..23},
// lanes 16-31 hold K {8..15,24..31} for row M = lane&15.
__device__ __forceinline__ int kpat(int j, int hi) {
  return (j < 8) ? (hi ? 8 + j : j) : (hi ? 16 + j : 8 + j);
}

// A fragment from a global fp32 row (this lane's M row), zero-pad K >= kmax.
__device__ __forceinline__ v16h loadA_g(const float* __restrict__ row, int lane, int kbase, int kmax) {
  int hi = lane >> 4;
  v16h a;
#pragma unroll
  for (int j = 0; j < 16; ++j) {
    int k = kbase + kpat(j, hi);
    a[j] = (k < kmax) ? (_Float16)row[k] : (_Float16)0.0f;
  }
  return a;
}

// A fragment from an LDS half row.
__device__ __forceinline__ v16h loadA_lds(const _Float16* row, int lane, int kbase) {
  int hi = lane >> 4;
  v16h a;
#pragma unroll
  for (int j = 0; j < 16; ++j) a[j] = row[kbase + kpat(j, hi)];
  return a;
}

// B fragment (32x16 tile of row-major [K,64] fp32 weights): N = lane&15, elements = K (kpat order).
__device__ __forceinline__ v16h loadB_g(const float* __restrict__ W, int lane,
                                        int kbase, int nb, int kmax) {
  v16h b;
  int hi = lane >> 4;
  const float* col = W + (unsigned)(nb * 16 + (lane & 15));
#pragma unroll
  for (int j = 0; j < 16; ++j) {
    int k = kbase + kpat(j, hi);
    b[j] = (k < kmax) ? (_Float16)col[(unsigned)k * 64u] : (_Float16)0.0f;
  }
  return b;
}

__device__ __forceinline__ float sspf(float x) {
  // softplus(x) - log(2), numerically stable
  return fmaxf(x, 0.0f) + log1pf(__expf(-fabsf(x))) - 0.69314718055994531f;
}

// ---------------- zero-init workspace ----------------
__global__ void k_zero(float* __restrict__ p, long n) {
  long i = (long)blockIdx.x * blockDim.x + threadIdx.x;
  long stride = (long)gridDim.x * blockDim.x;
  for (; i < n; i += stride) p[i] = 0.0f;
}

// ---------------- xs = x @ lin1_w  ([N,64] = [N,64]@[64,64]) ----------------
__global__ void k_xs(const float* __restrict__ x, const float* __restrict__ w,
                     float* __restrict__ xs, unsigned N, unsigned tiles) {
  int lane = threadIdx.x & 31;
  int wv = threadIdx.x >> 5;
  int m = lane & 15, hi = lane >> 4;

  v16h b[2][4];
#pragma unroll
  for (int kt = 0; kt < 2; ++kt)
#pragma unroll
    for (int nb = 0; nb < 4; ++nb) b[kt][nb] = loadB_g(w, lane, kt * 32, nb, 64);

  unsigned wstride = gridDim.x * 8u;
  for (unsigned tile = blockIdx.x * 8u + wv; tile < tiles; tile += wstride) {
    unsigned row0 = tile * 16u;
    unsigned myrow = row0 + m; if (myrow >= N) myrow = N - 1;
    const float* rb = x + (size_t)myrow * 64u;
    v16h a0 = loadA_g(rb, lane, 0, 64);
    v16h a1 = loadA_g(rb, lane, 32, 64);
    bool full = (row0 + 16u <= N);          // wave-uniform
#pragma unroll
    for (int nb = 0; nb < 4; ++nb) {
      v8f acc = {};
      acc = WMMA16(a0, b[0][nb], acc);
      acc = WMMA16(a1, b[1][nb], acc);
      unsigned col = (unsigned)(nb * 16 + m);
      if (full) {
#pragma unroll
        for (int r = 0; r < 8; ++r)
          xs[(row0 + (unsigned)(r + 8 * hi)) * 64u + col] = acc[r];
      } else {
#pragma unroll
        for (int r = 0; r < 8; ++r) {
          unsigned orow = row0 + (unsigned)(r + 8 * hi);
          if (orow < N) xs[orow * 64u + col] = acc[r];
        }
      }
    }
  }
}

// ---------------- fused edge pass: filter MLP -> cutoff -> gather -> scatter-add ----------------
// Each wave owns 16 edges; persistent over blocks of 128 edges. Per-wave LDS only.
template <int GG>
__global__ void k_edge(const float* __restrict__ ea, const float* __restrict__ ew,
                       const int* __restrict__ ei,
                       const float* __restrict__ mw1, const float* __restrict__ mb1,
                       const float* __restrict__ mw2, const float* __restrict__ mb2,
                       const float* __restrict__ xs, float* __restrict__ agg,
                       unsigned E, unsigned nblk) {
  __shared__ _Float16 sh_hidden[8][16 * 64];
  __shared__ float sh_C[8][16];
  __shared__ int sh_src[8][16];
  __shared__ int sh_dst[8][16];

  int lane = threadIdx.x & 31;
  int wv = threadIdx.x >> 5;
  int m = lane & 15, hi = lane >> 4;

  // Loop-invariant weight fragments + biases in registers (amortized over all tiles).
  v16h b1[2][4], b2[2][4];
  float bias1[4], bias2[4];
#pragma unroll
  for (int kt = 0; kt < 2; ++kt)
#pragma unroll
    for (int nb = 0; nb < 4; ++nb) {
      b1[kt][nb] = loadB_g(mw1, lane, kt * 32, nb, GG);
      b2[kt][nb] = loadB_g(mw2, lane, kt * 32, nb, 64);
    }
#pragma unroll
  for (int nb = 0; nb < 4; ++nb) {
    bias1[nb] = mb1[nb * 16 + m];
    bias2[nb] = mb2[nb * 16 + m];
  }

  for (unsigned blk = blockIdx.x; blk < nblk; blk += gridDim.x) {
    unsigned base = blk * 128u + (unsigned)(wv * 16);

    // per-edge scalars; out-of-range edges masked via C=0
    if (lane < 16) {
      unsigned e = base + (unsigned)lane;
      bool ok = (e < E);
      unsigned ec = ok ? e : (E - 1u);
      float d = ew[ec];
      sh_C[wv][lane] = ok ? 0.5f * (__cosf(d * 0.31415926535897931f) + 1.0f) : 0.0f;
      sh_src[wv][lane] = ei[ec];
      sh_dst[wv][lane] = ei[E + ec];
    }

    unsigned e = base + (unsigned)m; if (e >= E) e = E - 1u;
    const float* rb = ea + (size_t)e * (unsigned)GG;
    v16h a0 = loadA_g(rb, lane, 0, GG);
    v16h a1 = loadA_g(rb, lane, 32, GG);

    // GEMM1: hidden = ssp(ea @ mw1 + mb1) -> LDS (D-layout -> row-major transpose)
#pragma unroll
    for (int nb = 0; nb < 4; ++nb) {
      v8f acc = {};
      acc = WMMA16(a0, b1[0][nb], acc);
      acc = WMMA16(a1, b1[1][nb], acc);
      int col = nb * 16 + m;
#pragma unroll
      for (int r = 0; r < 8; ++r) {
        int row = r + 8 * hi;
        sh_hidden[wv][row * 64 + col] = (_Float16)sspf(acc[r] + bias1[nb]);
      }
    }
    LDS_FENCE();  // per-wave store->load ordering (DS ops in-order within wave)

    // GEMM2: W = (hidden @ mw2 + mb2) * C ; msg = xs[src] * W ; atomic scatter
    const _Float16* hrow = &sh_hidden[wv][m * 64];
    v16h h0 = loadA_lds(hrow, lane, 0);
    v16h h1 = loadA_lds(hrow, lane, 32);
#pragma unroll
    for (int nb = 0; nb < 4; ++nb) {
      v8f acc = {};
      acc = WMMA16(h0, b2[0][nb], acc);
      acc = WMMA16(h1, b2[1][nb], acc);
      unsigned col = (unsigned)(nb * 16 + m);
#pragma unroll
      for (int r = 0; r < 8; ++r) {
        int row = r + 8 * hi;
        float Wv = (acc[r] + bias2[nb]) * sh_C[wv][row];
        unsigned s = (unsigned)sh_src[wv][row];
        unsigned d = (unsigned)sh_dst[wv][row];
        float msg = Wv * xs[s * 64u + col];
        atomicAdd(&agg[d * 64u + col], msg);
      }
    }
    LDS_FENCE();  // don't overwrite sh_hidden before reads of this iteration retire
  }
}

// ---------------- node output: out = ssp(agg @ l2w + l2b) @ lw + lb ----------------
__global__ void k_out(const float* __restrict__ agg,
                      const float* __restrict__ l2w, const float* __restrict__ l2b,
                      const float* __restrict__ lw, const float* __restrict__ lb,
                      float* __restrict__ out, unsigned N, unsigned tiles) {
  __shared__ _Float16 sh_h[8][16 * 64];
  int lane = threadIdx.x & 31;
  int wv = threadIdx.x >> 5;
  int m = lane & 15, hi = lane >> 4;

  v16h bA[2][4], bB[2][4];
  float biasA[4], biasB[4];
#pragma unroll
  for (int kt = 0; kt < 2; ++kt)
#pragma unroll
    for (int nb = 0; nb < 4; ++nb) {
      bA[kt][nb] = loadB_g(l2w, lane, kt * 32, nb, 64);
      bB[kt][nb] = loadB_g(lw, lane, kt * 32, nb, 64);
    }
#pragma unroll
  for (int nb = 0; nb < 4; ++nb) {
    biasA[nb] = l2b[nb * 16 + m];
    biasB[nb] = lb[nb * 16 + m];
  }

  unsigned wstride = gridDim.x * 8u;
  for (unsigned tile = blockIdx.x * 8u + wv; tile < tiles; tile += wstride) {
    unsigned row0 = tile * 16u;
    unsigned myrow = row0 + m; if (myrow >= N) myrow = N - 1;
    const float* rb = agg + (size_t)myrow * 64u;
    v16h a0 = loadA_g(rb, lane, 0, 64);
    v16h a1 = loadA_g(rb, lane, 32, 64);
#pragma unroll
    for (int nb = 0; nb < 4; ++nb) {
      v8f acc = {};
      acc = WMMA16(a0, bA[0][nb], acc);
      acc = WMMA16(a1, bA[1][nb], acc);
      int col = nb * 16 + m;
#pragma unroll
      for (int r = 0; r < 8; ++r) {
        int row = r + 8 * hi;
        sh_h[wv][row * 64 + col] = (_Float16)sspf(acc[r] + biasA[nb]);
      }
    }
    LDS_FENCE();
    const _Float16* hrow = &sh_h[wv][m * 64];
    v16h h0 = loadA_lds(hrow, lane, 0);
    v16h h1 = loadA_lds(hrow, lane, 32);
    bool full = (row0 + 16u <= N);          // wave-uniform
#pragma unroll
    for (int nb = 0; nb < 4; ++nb) {
      v8f acc = {};
      acc = WMMA16(h0, bB[0][nb], acc);
      acc = WMMA16(h1, bB[1][nb], acc);
      unsigned col = (unsigned)(nb * 16 + m);
      if (full) {
#pragma unroll
        for (int r = 0; r < 8; ++r)
          out[(row0 + (unsigned)(r + 8 * hi)) * 64u + col] = acc[r] + biasB[nb];
      } else {
#pragma unroll
        for (int r = 0; r < 8; ++r) {
          unsigned orow = row0 + (unsigned)(r + 8 * hi);
          if (orow < N) out[orow * 64u + col] = acc[r] + biasB[nb];
        }
      }
    }
    LDS_FENCE();
  }
}

static inline int imin(int a, int b) { return a < b ? a : b; }

extern "C" void kernel_launch(void* const* d_in, const int* in_sizes, int n_in,
                              void* d_out, int out_size, void* d_ws, size_t ws_size,
                              hipStream_t stream) {
  const float* x0   = (const float*)d_in[0];
  const float* x1   = (const float*)d_in[1];
  const int*   ei0  = (const int*)d_in[2];
  const int*   ei1  = (const int*)d_in[3];
  const float* ew0  = (const float*)d_in[4];
  const float* ew1  = (const float*)d_in[5];
  const float* ea0  = (const float*)d_in[6];
  const float* ea1  = (const float*)d_in[7];
  const float* mw1  = (const float*)d_in[8];
  const float* mb1  = (const float*)d_in[9];
  const float* mw2  = (const float*)d_in[10];
  const float* mb2  = (const float*)d_in[11];
  const float* l1w0 = (const float*)d_in[12];
  const float* l2w0 = (const float*)d_in[13];
  const float* l2b0 = (const float*)d_in[14];
  const float* l1w1 = (const float*)d_in[15];
  const float* l2w1 = (const float*)d_in[16];
  const float* l2b1 = (const float*)d_in[17];
  const float* lw   = (const float*)d_in[18];
  const float* lb   = (const float*)d_in[19];

  const unsigned N = (unsigned)(in_sizes[0] / 64);  // H = F = 64 fixed by WMMA tiling
  const unsigned E = (unsigned)in_sizes[4];
  const int      G = in_sizes[8] / 64;              // mlp_w1 is [G,64]

  float* xs0  = (float*)d_ws;
  float* xs1  = xs0  + (size_t)N * 64;
  float* agg0 = xs1  + (size_t)N * 64;
  float* agg1 = agg0 + (size_t)N * 64;

  float* out0 = (float*)d_out;
  float* out1 = out0 + (size_t)N * 64;

  // 1. zero the aggregation buffers (workspace is poisoned)
  {
    long n = 2L * (long)N * 64;
    int blocks = imin((int)((n + 255) / 256), 2048);
    k_zero<<<blocks, 256, 0, stream>>>(agg0, n);
  }

  // 2. xs = x @ lin1_w for both edge types
  unsigned tiles = (N + 15u) / 16u;
  int nodeBlocks = imin((int)((tiles + 7u) / 8u), 512);
  k_xs<<<nodeBlocks, 256, 0, stream>>>(x0, l1w0, xs0, N, tiles);
  k_xs<<<nodeBlocks, 256, 0, stream>>>(x1, l1w1, xs1, N, tiles);

  // 3. fused edge passes (filter MLP + cutoff + gather + atomic scatter-sum)
  {
    unsigned nblk = (E + 127u) / 128u;
    int blocks = imin((int)nblk, 1024);
    if (G == 50) {
      k_edge<50><<<blocks, 256, 0, stream>>>(ea0, ew0, ei0, mw1, mb1, mw2, mb2, xs0, agg0, E, nblk);
      k_edge<50><<<blocks, 256, 0, stream>>>(ea1, ew1, ei1, mw1, mb1, mw2, mb2, xs1, agg1, E, nblk);
    } else {
      k_edge<64><<<blocks, 256, 0, stream>>>(ea0, ew0, ei0, mw1, mb1, mw2, mb2, xs0, agg0, E, nblk);
      k_edge<64><<<blocks, 256, 0, stream>>>(ea1, ew1, ei1, mw1, mb1, mw2, mb2, xs1, agg1, E, nblk);
    }
  }

  // 4. node output: out = ssp(agg @ lin2 + b2) @ lin_w + lin_b
  k_out<<<nodeBlocks, 256, 0, stream>>>(agg0, l2w0, l2b0, lw, lb, out0, N, tiles);
  k_out<<<nodeBlocks, 256, 0, stream>>>(agg1, l2w1, l2b1, lw, lb, out1, N, tiles);
}
